// DCNv3_Block_41618233098510
// MI455X (gfx1250) — compile-verified, matrix operand order checked by reading, not run on hardware
//
#include <hip/hip_runtime.h>
#include <math.h>

// ---------------------------------------------------------------------------
// DCNv3 block for MI455X (gfx1250, wave32, WMMA + async LDS staging).
// Activations flow in bf16 between matmul stages; weights pre-transposed
// N-major so both WMMA operands load as 16B chunks (global b128 / ds b128).
// ---------------------------------------------------------------------------

typedef __attribute__((ext_vector_type(16))) __bf16 v16bf;
typedef __attribute__((ext_vector_type(8)))  __bf16 v8bfv;
typedef __attribute__((ext_vector_type(8)))  float  v8f;
typedef int v4i_g __attribute__((vector_size(4 * sizeof(int))));

#define NB   4
#define HH   64
#define WW   64
#define CC_  256
#define LL   4096            // HH*WW
#define MM   16384           // NB*LL
#define GG   8
#define GCC  32
#define PP   9
#define HO2  32
#define WO2  32
#define MM2  4096            // NB*HO2*WO2
#define EPSV 1e-5f
#define CSTR 264             // LDS column stride (bank-conflict padding)

#if defined(__has_builtin)
#if __has_builtin(__builtin_amdgcn_global_load_async_to_lds_b128) && \
    __has_builtin(__builtin_amdgcn_s_wait_asynccnt)
#define HAVE_ASYNC 1
#endif
#endif
#ifndef HAVE_ASYNC
#define HAVE_ASYNC 0
#endif
// builtin signature (from hipcc diagnostic): (global int4*, local int4*, imm, imm)
#define GPTR(p) ((__attribute__((address_space(1))) v4i_g*)(p))
#define LPTR(p) ((__attribute__((address_space(3))) v4i_g*)(p))

static __device__ __forceinline__ unsigned short bf_of(float f) {
  union { float f; unsigned int u; } v; v.f = f;
  return (unsigned short)((v.u + 0x7FFFu + ((v.u >> 16) & 1u)) >> 16);
}

// ---------------- weight conversion f32 [K,N] -> bf16 transposed [Ndst,K] ---
__global__ void cvt_bf16_t(const float* __restrict__ src, unsigned short* __restrict__ dst,
                           int K, int Nsrc, int total) {
  int t = blockIdx.x * blockDim.x + threadIdx.x;
  if (t >= total) return;
  int n = t / K, k = t % K;
  dst[t] = (n < Nsrc) ? bf_of(src[(long)k * Nsrc + n]) : (unsigned short)0;
}

// ---------------- NCHW -> NHWC (f32 + bf16 mirror) -------------------------
__global__ void nchw2nhwc(const float* __restrict__ X, float* __restrict__ Yf,
                          unsigned short* __restrict__ Yb, int total) {
  int t = blockIdx.x * blockDim.x + threadIdx.x;
  if (t >= total) return;
  int c = t & 255;
  int idx = t >> 8;
  int n = idx >> 12;
  int hw = idx & 4095;
  float v = X[(((long)n * CC_ + c) << 12) + hw];
  Yf[t] = v;
  Yb[t] = bf_of(v);
}

// ---- stage 16 cols x 256 K bf16 panel into LDS (async global->LDS) --------
static __device__ __forceinline__ void stage_panel_t(const unsigned short* __restrict__ gbase,
                                                     long Kd, unsigned short* lds) {
  for (int i = threadIdx.x; i < 512; i += 128) {     // 512 x 16B chunks
    int col = i >> 5, ck = (i & 31) << 3;            // 8 bf16 per chunk
    const unsigned short* gsrc = gbase + (long)col * Kd + ck;
    unsigned short* ldst = lds + col * CSTR + ck;
#if HAVE_ASYNC
    __builtin_amdgcn_global_load_async_to_lds_b128(GPTR(gsrc), LPTR(ldst), 0, 0);
#else
    *(v8bfv*)ldst = *(const v8bfv*)gsrc;
#endif
  }
#if HAVE_ASYNC
  __builtin_amdgcn_s_wait_asynccnt(0);
#endif
  __syncthreads();
}

// fragment: two contiguous 16B bf16 runs (works for LDS panel or global row)
static __device__ __forceinline__ v16bf frag16(const unsigned short* p, int k0, int khalf) {
  v8bfv lo = *(const v8bfv*)(p + k0 + khalf);
  v8bfv hi = *(const v8bfv*)(p + k0 + 16 + khalf);
  return __builtin_shufflevector(lo, hi, 0, 1, 2, 3, 4, 5, 6, 7,
                                 8, 9, 10, 11, 12, 13, 14, 15);
}

static __device__ __forceinline__ v16bf zero_frag() {
  union { unsigned short s[16]; v16bf v; } u;
#pragma unroll
  for (int j = 0; j < 16; ++j) u.s[j] = 0;
  return u.v;
}

// ---------------- WMMA GEMM: C[M,Np] = A[M,256](bf16) * Wt + bias ----------
// Wt transposed: [Npad][256] bf16. Block = 4 waves, tile 64(M) x 16(N).
__global__ void gemm_wmma(const unsigned short* __restrict__ A,
                          const unsigned short* __restrict__ Wt,
                          const float* __restrict__ bias, float* __restrict__ Cm,
                          int Np, int Nreal) {
  __shared__ __align__(16) unsigned short panel[16 * CSTR];
  int wave = threadIdx.x >> 5, lane = threadIdx.x & 31;
  int tn = blockIdx.x << 4;                       // gridDim.x = Np/16
  int tm = (blockIdx.y << 6) + (wave << 4);       // gridDim.y = M/64

  stage_panel_t(Wt + (long)tn * 256, 256, panel);

  int row = lane & 15;
  int khalf = (lane & 16) ? 8 : 0;
  const unsigned short* Arow = A + (long)(tm + row) * 256;
  const unsigned short* colp = panel + row * CSTR;
  __builtin_prefetch(Arow, 0, 3);                 // global_prefetch_b8
  v8f acc = {0.f, 0.f, 0.f, 0.f, 0.f, 0.f, 0.f, 0.f};
#pragma unroll
  for (int k0 = 0; k0 < 256; k0 += 32) {
    v16bf a = frag16(Arow, k0, khalf);
    v16bf b = frag16(colp, k0, khalf);
    acc = __builtin_amdgcn_wmma_f32_16x16x32_bf16(false, a, false, b, (short)0, acc, false, false);
  }
  int col = tn + row;
  float bv = (col < Nreal) ? bias[col] : 0.f;
  int rb = (lane & 16) ? 8 : 0;
#pragma unroll
  for (int e = 0; e < 8; ++e)
    Cm[(long)(tm + rb + e) * Np + col] = acc[e] + bv;
}

// ---------------- implicit-GEMM 3x3 conv (pad 1), Cin=Cout=256, bf16 in ----
__global__ void conv3x3_wmma(const unsigned short* __restrict__ X,
                             const unsigned short* __restrict__ Wt,
                             const float* __restrict__ bias, float* __restrict__ Y,
                             int Hi, int Wi, int Ho, int Wo, int stride) {
  __shared__ __align__(16) unsigned short panel[16 * CSTR];
  int wave = threadIdx.x >> 5, lane = threadIdx.x & 31;
  int tn = blockIdx.x << 4;                       // gridDim.x = 16
  int tm = (blockIdx.y << 6) + (wave << 4);       // gridDim.y = M/64
  int row = lane & 15;
  int khalf = (lane & 16) ? 8 : 0;
  int m = tm + row;
  int n = m / (Ho * Wo);
  int rem = m % (Ho * Wo);
  int ho = rem / Wo, wo = rem % Wo;
  const unsigned short* colp = panel + row * CSTR;
  v8f acc = {0.f, 0.f, 0.f, 0.f, 0.f, 0.f, 0.f, 0.f};
  for (int tap = 0; tap < 9; ++tap) {
    int ky = tap / 3, kx = tap % 3;
    __syncthreads();                              // previous panel fully consumed
    stage_panel_t(Wt + (long)tn * 2304 + tap * 256, 2304, panel);
    int hin = ho * stride + ky - 1;
    int win = wo * stride + kx - 1;
    bool valid = (hin >= 0) & (hin < Hi) & (win >= 0) & (win < Wi);
    const unsigned short* src = X + (((long)n * Hi + (valid ? hin : 0)) * Wi + (valid ? win : 0)) * 256;
#pragma unroll
    for (int kc = 0; kc < 256; kc += 32) {
      v16bf a = frag16(src, kc, khalf);
      if (!valid) a = zero_frag();
      v16bf b = frag16(colp, kc, khalf);
      acc = __builtin_amdgcn_wmma_f32_16x16x32_bf16(false, a, false, b, (short)0, acc, false, false);
    }
  }
  int col = tn + row;
  float bv = bias[col];
  int rb = (lane & 16) ? 8 : 0;
#pragma unroll
  for (int e = 0; e < 8; ++e)
    Y[(long)(tm + rb + e) * 256 + col] = acc[e] + bv;
}

// ---------------- depthwise 3x3 + LayerNorm(C) + exact GELU -> bf16 --------
__global__ void dwln_gelu(const float* __restrict__ XS, const float* __restrict__ dww,
                          const float* __restrict__ dwb, const float* __restrict__ lng,
                          const float* __restrict__ lnb, unsigned short* __restrict__ T) {
  __shared__ float s1[256], s2[256];
  int m = blockIdx.x, c = threadIdx.x;
  int n = m >> 12, h = (m >> 6) & 63, w = m & 63;
  float acc = dwb[c];
  for (int ky = 0; ky < 3; ++ky) {
    int hh = h + ky - 1;
    if (hh < 0 || hh >= HH) continue;
    for (int kx = 0; kx < 3; ++kx) {
      int ww = w + kx - 1;
      if (ww < 0 || ww >= WW) continue;
      acc += XS[(((long)n << 12) + hh * WW + ww) * CC_ + c] * dww[(ky * 3 + kx) * CC_ + c];
    }
  }
  s1[c] = acc; s2[c] = acc * acc;
  __syncthreads();
  for (int off = 128; off > 0; off >>= 1) {
    if (c < off) { s1[c] += s1[c + off]; s2[c] += s2[c + off]; }
    __syncthreads();
  }
  float mu = s1[0] * (1.f / 256.f);
  float var = s2[0] * (1.f / 256.f) - mu * mu;
  float t = (acc - mu) * rsqrtf(var + EPSV) * lng[c] + lnb[c];
  t = 0.5f * t * (1.f + erff(t * 0.70710678118f));
  T[(long)m * CC_ + c] = bf_of(t);
}

// ---------------- softmax over P=9 per (pixel, group); rows padded to 80 ----
__global__ void softmax9(float* __restrict__ MB, int total) {
  int t = blockIdx.x * blockDim.x + threadIdx.x;
  if (t >= total) return;
  int m = t >> 3, g = t & 7;
  float* p = MB + (long)m * 80 + g * 9;
  float mx = p[0];
#pragma unroll
  for (int i = 1; i < 9; ++i) mx = fmaxf(mx, p[i]);
  float e[9]; float s = 0.f;
#pragma unroll
  for (int i = 0; i < 9; ++i) { e[i] = __expf(p[i] - mx); s += e[i]; }
  float inv = 1.f / s;
#pragma unroll
  for (int i = 0; i < 9; ++i) p[i] = e[i] * inv;
}

// ---------------- deformable bilinear gather + aggregate -> bf16 -----------
__global__ void dcn_sample(const float* __restrict__ XP, const float* __restrict__ OFF,
                           const float* __restrict__ MB, unsigned short* __restrict__ AGG) {
  __shared__ float soff[144];
  __shared__ float smk[72];
  int m = blockIdx.x, t = threadIdx.x;
  if (t < 144) soff[t] = OFF[(long)m * 144 + t];
  else if (t < 216) smk[t - 144] = MB[(long)m * 80 + (t - 144)];
  __syncthreads();
  int g = t >> 5, cc = t & 31;
  int n = m >> 12, h = (m >> 6) & 63, w = m & 63;
  const float* img = XP + ((long)n << 12) * CC_ + g * GCC + cc;
  float acc = 0.f;
  for (int p = 0; p < 9; ++p) {
    float ox = soff[(g * 9 + p) * 2];
    float oy = soff[(g * 9 + p) * 2 + 1];
    float mv = smk[g * 9 + p];
    float px = 1.f + (float)w + (float)(p / 3 - 1) + ox;
    float py = 1.f + (float)h + (float)(p % 3 - 1) + oy;
    float x0 = floorf(px), y0 = floorf(py);
    float fx = px - x0, fy = py - y0;
#pragma unroll
    for (int dx = 0; dx < 2; ++dx) {
#pragma unroll
      for (int dy = 0; dy < 2; ++dy) {
        float xi = x0 + (float)dx, yi = y0 + (float)dy;
        float wgt = (dx ? fx : 1.f - fx) * (dy ? fy : 1.f - fy) * mv;
        if (xi >= 1.f && xi <= 64.f && yi >= 1.f && yi <= 64.f) {
          int ix = (int)xi - 1, iy = (int)yi - 1;
          acc += wgt * img[(long)(iy * WW + ix) * CC_];
        }
      }
    }
  }
  AGG[(long)m * CC_ + g * GCC + cc] = bf_of(acc);
}

// ---------------- GroupNorm(32) stats --------------------------------------
__global__ void gn_stats(const float* __restrict__ X, int S, float* __restrict__ st) {
  __shared__ float s1[256], s2[256];
  int b = blockIdx.x;
  int n = b >> 5, gr = b & 31;
  int cnt = S * 8;
  float sum = 0.f, sq = 0.f;
  for (int i = threadIdx.x; i < cnt; i += 256) {
    int s = i >> 3, ch = (gr << 3) + (i & 7);
    float v = X[((long)n * S + s) * CC_ + ch];
    sum += v; sq += v * v;
  }
  s1[threadIdx.x] = sum; s2[threadIdx.x] = sq;
  __syncthreads();
  for (int off = 128; off > 0; off >>= 1) {
    if (threadIdx.x < off) { s1[threadIdx.x] += s1[threadIdx.x + off]; s2[threadIdx.x] += s2[threadIdx.x + off]; }
    __syncthreads();
  }
  if (threadIdx.x == 0) {
    float mu = s1[0] / (float)cnt;
    float var = s2[0] / (float)cnt - mu * mu;
    st[b * 2] = mu;
    st[b * 2 + 1] = rsqrtf(var + EPSV);
  }
}

// GN apply + SiLU, bf16 output (feeds the WMMA convs)
__global__ void gn_apply_silu_bf(const float* __restrict__ X, const float* __restrict__ st,
                                 const float* __restrict__ gam, const float* __restrict__ bet,
                                 unsigned short* __restrict__ Y, int S, int total) {
  int t = blockIdx.x * blockDim.x + threadIdx.x;
  if (t >= total) return;
  int c = t & 255;
  int idx = t >> 8;
  int n = idx / S;
  int gr = c >> 3;
  float mu = st[(n * 32 + gr) * 2], rs = st[(n * 32 + gr) * 2 + 1];
  float v = (X[t] - mu) * rs * gam[c] + bet[c];
  v = v / (1.f + __expf(-v));
  Y[t] = bf_of(v);
}

// final GN (no act) + NHWC -> NCHW into d_out (f32)
__global__ void gn_apply_nchw(const float* __restrict__ X, const float* __restrict__ st,
                              const float* __restrict__ gam, const float* __restrict__ bet,
                              float* __restrict__ Y, int Hs, int Ws, int total) {
  int t = blockIdx.x * blockDim.x + threadIdx.x;
  if (t >= total) return;
  int c = t & 255;
  int idx = t >> 8;
  int S = Hs * Ws;
  int n = idx / S, sp = idx % S;
  int gr = c >> 3;
  float mu = st[(n * 32 + gr) * 2], rs = st[(n * 32 + gr) * 2 + 1];
  float v = (X[t] - mu) * rs * gam[c] + bet[c];
  Y[(((long)n * CC_ + c) * Hs + sp / Ws) * Ws + sp % Ws] = v;
}

// ---------------------------------------------------------------------------
extern "C" void kernel_launch(void* const* d_in, const int* in_sizes, int n_in,
                              void* d_out, int out_size, void* d_ws, size_t ws_size,
                              hipStream_t stream) {
  (void)in_sizes; (void)n_in; (void)out_size; (void)ws_size;
  const float* x      = (const float*)d_in[0];
  const float* w_in   = (const float*)d_in[1];
  const float* b_in   = (const float*)d_in[2];
  const float* dw_w   = (const float*)d_in[3];
  const float* dw_b   = (const float*)d_in[4];
  const float* ln_g   = (const float*)d_in[5];
  const float* ln_b   = (const float*)d_in[6];
  const float* w_off  = (const float*)d_in[7];
  const float* b_off  = (const float*)d_in[8];
  const float* w_mask = (const float*)d_in[9];
  const float* b_mask = (const float*)d_in[10];
  const float* w_out  = (const float*)d_in[11];
  const float* b_out  = (const float*)d_in[12];
  const float* gn1_g  = (const float*)d_in[13];
  const float* gn1_b  = (const float*)d_in[14];
  const float* c1_w   = (const float*)d_in[15];
  const float* c1_b   = (const float*)d_in[16];
  const float* gn2_g  = (const float*)d_in[17];
  const float* gn2_b  = (const float*)d_in[18];
  const float* dn_w   = (const float*)d_in[19];
  const float* dn_b   = (const float*)d_in[20];
  const float* gn3_g  = (const float*)d_in[21];
  const float* gn3_b  = (const float*)d_in[22];

  const long MC = (long)MM * CC_;
  float* ws   = (float*)d_ws;
  float* xs   = ws;                        // NHWC input f32 (later: conv1 out h2)
  float* xpj  = xs + MC;                   // x_proj f32 (sampling source)
  float* yb   = xpj + MC;                  // output-proj result f32
  float* ofb  = yb + MC;                   // offsets 16384x144 (later: h4 f32)
  float* mkb  = ofb + (long)MM * 144;      // mask, row-stride 80
  float* st   = mkb + (long)MM * 80;       // GN stats
  unsigned short* xs_bf  = (unsigned short*)(st + 256);     // input bf16
  unsigned short* tb_bf  = xs_bf + MC;                      // t bf16 (later: h3 bf16)
  unsigned short* ag_bf  = tb_bf + MC;                      // agg bf16 (later: h1 bf16)
  unsigned short* w_in_bf  = ag_bf + MC;                    // [256][256]
  unsigned short* w_off_bf = w_in_bf  + 65536;              // [144][256]
  unsigned short* w_msk_bf = w_off_bf + 36864;              // [80][256]
  unsigned short* w_out_bf = w_msk_bf + 20480;              // [256][256]
  unsigned short* w_c1_bf  = w_out_bf + 65536;              // [256][2304]
  unsigned short* w_dn_bf  = w_c1_bf  + 589824;             // [256][2304]

  // transposed bf16 weights (N-major, K-contiguous rows)
  cvt_bf16_t<<<256, 256, 0, stream>>>(w_in, w_in_bf, 256, 256, 65536);
  cvt_bf16_t<<<144, 256, 0, stream>>>(w_off, w_off_bf, 256, 144, 36864);
  cvt_bf16_t<<<80, 256, 0, stream>>>(w_mask, w_msk_bf, 256, 72, 20480);
  cvt_bf16_t<<<256, 256, 0, stream>>>(w_out, w_out_bf, 256, 256, 65536);
  cvt_bf16_t<<<2304, 256, 0, stream>>>(c1_w, w_c1_bf, 2304, 256, 589824);
  cvt_bf16_t<<<2304, 256, 0, stream>>>(dn_w, w_dn_bf, 2304, 256, 589824);

  nchw2nhwc<<<(int)(MC / 256), 256, 0, stream>>>(x, xs, xs_bf, (int)MC);

  // input_proj (bf16 A x bf16 W -> f32)
  gemm_wmma<<<dim3(16, MM / 64), 128, 0, stream>>>(xs_bf, w_in_bf, b_in, xpj, 256, 256);

  // depthwise + LN + GELU -> bf16
  dwln_gelu<<<MM, 256, 0, stream>>>(xs, dw_w, dw_b, ln_g, ln_b, tb_bf);

  // offset / mask projections
  gemm_wmma<<<dim3(9, MM / 64), 128, 0, stream>>>(tb_bf, w_off_bf, b_off, ofb, 144, 144);
  gemm_wmma<<<dim3(5, MM / 64), 128, 0, stream>>>(tb_bf, w_msk_bf, b_mask, mkb, 80, 72);

  softmax9<<<(MM * GG) / 256, 256, 0, stream>>>(mkb, MM * GG);

  // deformable sampling -> bf16 agg
  dcn_sample<<<MM, 256, 0, stream>>>(xpj, ofb, mkb, ag_bf);

  // output_proj
  gemm_wmma<<<dim3(16, MM / 64), 128, 0, stream>>>(ag_bf, w_out_bf, b_out, yb, 256, 256);

  // GN1 + SiLU -> h1 bf16 (reuses ag_bf)
  gn_stats<<<128, 256, 0, stream>>>(yb, LL, st);
  gn_apply_silu_bf<<<(int)(MC / 256), 256, 0, stream>>>(yb, st, gn1_g, gn1_b, ag_bf, LL, (int)MC);

  // conv1 -> h2 f32 (reuses xs)
  conv3x3_wmma<<<dim3(16, MM / 64), 128, 0, stream>>>(ag_bf, w_c1_bf, c1_b, xs, HH, WW, HH, WW, 1);

  // GN2 + SiLU -> h3 bf16 (reuses tb_bf)
  gn_stats<<<128, 256, 0, stream>>>(xs, LL, st);
  gn_apply_silu_bf<<<(int)(MC / 256), 256, 0, stream>>>(xs, st, gn2_g, gn2_b, tb_bf, LL, (int)MC);

  // down conv stride 2 -> h4 f32 (reuses ofb)
  conv3x3_wmma<<<dim3(16, MM2 / 64), 128, 0, stream>>>(tb_bf, w_dn_bf, dn_b, ofb, HH, WW, HO2, WO2, 2);

  // GN3 + NHWC->NCHW into d_out
  gn_stats<<<128, 256, 0, stream>>>(ofb, HO2 * WO2, st);
  gn_apply_nchw<<<(MM2 * CC_) / 256, 256, 0, stream>>>(ofb, st, gn3_g, gn3_b,
                                                       (float*)d_out, HO2, WO2,
                                                       MM2 * CC_);
}
